// PhysicsInformedODE_13469017440461
// MI455X (gfx1250) — matrix operand order repaired and checked by reading, not compiled, not run on hardware
//
#include <hip/hip_runtime.h>
#include <hip/hip_bf16.h>

// ---------------------------------------------------------------------------
// PhysicsInformedODE on MI455X (gfx1250): f16 WMMA MLPs + VALU softmax blend.
// B=262144, D=64, P=8, Z=8, E=16.
// ---------------------------------------------------------------------------

typedef __attribute__((ext_vector_type(16))) _Float16 v16h_t;
typedef __attribute__((ext_vector_type(8)))  _Float16 v8h_t;
typedef __attribute__((ext_vector_type(8)))  float    v8f_t;

union V16H { v16h_t v; v8h_t h[2]; };

#define BB 262144
#define DD 64
#define PP 8
#define ZZ 8
#define EE 16

// wB tile table: tile = base + kc*NT + nt, each tile = 512 halves (32 lanes x 16)
#define T_PE1 0    // 1 kc x 4 nt
#define T_PE2 4    // 2 kc x 4 nt
#define T_FN1 12   // 5 kc x 4 nt (K 144 padded to 160)
#define T_FN2 32   // 2 kc x 2 nt
#define T_FN3 36   // 1 kc x 4 nt
#define NTILES 40

// ---------------------------------------------------------------------------
// Prep: adjacency bitmask, zone norms, direction table, f16 B-layout weights.
// ---------------------------------------------------------------------------
__global__ void ode_prep(const int* __restrict__ edge,
                         const float* __restrict__ ze,
                         const float* __restrict__ pe_w1,
                         const float* __restrict__ pe_w2,
                         const float* __restrict__ fn_w1,
                         const float* __restrict__ fn_w2,
                         const float* __restrict__ fn_w3,
                         float* __restrict__ dir,
                         unsigned* __restrict__ adjbits,
                         float* __restrict__ zn2,
                         _Float16* __restrict__ wB) {
  const int tid = threadIdx.x;

  if (tid < ZZ) {
    unsigned bits = 1u << tid;                   // self loop
    for (int e = 0; e < EE; ++e) {
      int a = edge[e];                           // edge_index row 0
      int b = edge[EE + e];                      // edge_index row 1
      if (a == tid) bits |= 1u << b;
      if (b == tid) bits |= 1u << a;
    }
    adjbits[tid] = bits;
    float s = 0.f;
    for (int d = 0; d < DD; ++d) { float v = ze[tid * DD + d]; s += v * v; }
    zn2[tid] = s;
  }

  if (tid < ZZ * ZZ) {
    int i = tid >> 3, j = tid & 7;
    float n2 = 0.f;
    for (int d = 0; d < DD; ++d) {
      float df = ze[j * DD + d] - ze[i * DD + d];
      n2 += df * df;
    }
    float nrm = sqrtf(n2);
    float sc = (nrm > 1e-6f) ? (1.0f / nrm) : 1.0f;   // zero diff stays zero
    for (int d = 0; d < DD; ++d)
      dir[(i * ZZ + j) * DD + d] = (ze[j * DD + d] - ze[i * DD + d]) * sc;
  }

  // Weights -> f16, WMMA B-operand lane layout:
  // within a tile, index = lane*16 + i ; K = kc*32 + (i>>3)*16 + (lane>>4)*8 + (i&7) ; N = nt*16 + (lane&15)
  for (int idx = tid; idx < NTILES * 512; idx += blockDim.x) {
    int t    = idx >> 9;
    int pos  = idx & 511;
    int lane = pos >> 4;
    int i    = pos & 15;
    int kk   = ((i >> 3) << 4) + ((lane >> 4) << 3) + (i & 7);
    int n    = lane & 15;
    const float* src; int kc, nt, Kact, N;
    if (t < T_PE2)      { src = pe_w1; kc = 0;            nt = t;           Kact = 8;   N = 64; }
    else if (t < T_FN1) { src = pe_w2; kc = (t - 4) >> 2; nt = (t - 4) & 3; Kact = 64;  N = 64; }
    else if (t < T_FN2) { src = fn_w1; kc = (t - 12) >> 2; nt = (t - 12) & 3; Kact = 144; N = 64; }
    else if (t < T_FN3) { src = fn_w2; kc = (t - 32) >> 1; nt = (t - 32) & 1; Kact = 64;  N = 32; }
    else                { src = fn_w3; kc = 0;            nt = t - 36;      Kact = 32;  N = 64; }
    int K = kc * 32 + kk;
    float v = (K < Kact) ? src[K * N + nt * 16 + n] : 0.0f;
    wB[idx] = (_Float16)v;
  }
}

// ---------------------------------------------------------------------------
// WMMA fragment helpers (layouts per CDNA5 ISA 7.12.2).
// ---------------------------------------------------------------------------
__device__ __forceinline__ v16h_t lds_load_A(const _Float16* buf, int ldh, int kbase, int lane) {
  // 16-bit A 16x32: lane m=lane&15, half h=lane>>4; VGPR0-3: K=h*8+0..7, VGPR4-7: K=16+h*8+0..7
  int m = lane & 15, hh = lane >> 4;
  const _Float16* p = buf + m * ldh + kbase + hh * 8;
  V16H r;
  r.h[0] = *(const v8h_t*)(p);
  r.h[1] = *(const v8h_t*)(p + 16);
  return r.v;
}

__device__ __forceinline__ v16h_t g_load_B(const _Float16* wB, int tile, int lane) {
  // pre-swizzled: lane's 16 halves contiguous
  const _Float16* p = wB + tile * 512 + lane * 16;
  V16H r;
  r.h[0] = *(const v8h_t*)(p);
  r.h[1] = *(const v8h_t*)(p + 8);
  return r.v;
}

__device__ __forceinline__ v8f_t bias_c(const float* bias, int nbase, int lane) {
  float b = bias[nbase + (lane & 15)];
  v8f_t c;
#pragma unroll
  for (int r = 0; r < 8; ++r) c[r] = b;
  return c;
}

__device__ __forceinline__ void store_C_f16(v8f_t c, _Float16* buf, int ldh, int nbase,
                                            int lane, bool relu) {
  // C/D f32 16x16: VGPR r, lanes 0-15 -> M=r, lanes 16-31 -> M=r+8; N=lane&15
  int n = nbase + (lane & 15);
  int mb = (lane >> 4) * 8;
#pragma unroll
  for (int r = 0; r < 8; ++r) {
    float x = c[r];
    if (relu) x = fmaxf(x, 0.f);
    buf[(mb + r) * ldh + n] = (_Float16)x;
  }
}

#define WMMA(a, b, c) \
  __builtin_amdgcn_wmma_f32_16x16x32_f16(false, (a), false, (b), (short)0, (c), false, false)

// ---------------------------------------------------------------------------
// Main kernel: 128 threads = 4 waves, 16 rows per wave.
// ---------------------------------------------------------------------------
#define WAVES 4

__global__ __launch_bounds__(128) void ode_main(
    const float* __restrict__ t_in,
    const float* __restrict__ state,
    const float* __restrict__ pa,
    const float* __restrict__ ze,
    const float* __restrict__ te_w1, const float* __restrict__ te_b1,
    const float* __restrict__ te_w2, const float* __restrict__ te_b2,
    const float* __restrict__ pe_b1, const float* __restrict__ pe_b2,
    const float* __restrict__ fn_b1, const float* __restrict__ fn_b2,
    const float* __restrict__ fn_b3,
    const float* __restrict__ dir,
    const unsigned* __restrict__ adjbits,
    const float* __restrict__ zn2,
    const _Float16* __restrict__ wB,
    float* __restrict__ out) {
  __shared__ __align__(16) _Float16 s_abuf[WAVES][16][160];  // comb [state|p_enc|t_enc|pad]
  __shared__ __align__(16) _Float16 s_hbuf[WAVES][16][64];   // hidden
  __shared__ __align__(16) _Float16 s_ubuf[WAVES][16][64];   // union: pa(16x32) / fn2out(16x32) / v(16x64)

  const int wv   = threadIdx.x >> 5;
  const int lane = threadIdx.x & 31;
  const long r0  = (long)blockIdx.x * (WAVES * 16) + (long)wv * 16;

  _Float16* abuf = &s_abuf[wv][0][0];
  _Float16* hbuf = &s_hbuf[wv][0][0];
  _Float16* ubuf = &s_ubuf[wv][0][0];

  // ---- stage state (f32 -> f16) into abuf cols 0..63 -----------------------
#pragma unroll 2
  for (int idx = lane; idx < 256; idx += 32) {
    int row = idx >> 4, c4 = (idx & 15) * 4;
    const float4 s4 = *(const float4*)(state + (r0 + row) * DD + c4);
    _Float16* d = abuf + row * 160 + c4;
    d[0] = (_Float16)s4.x; d[1] = (_Float16)s4.y;
    d[2] = (_Float16)s4.z; d[3] = (_Float16)s4.w;
  }

  // ---- stage person_attrs into ubuf (16x32, zero padded K 8..31) -----------
  for (int idx = lane; idx < 512; idx += 32) ubuf[idx] = (_Float16)0.f;
  __builtin_amdgcn_wave_barrier();
  for (int idx = lane; idx < 128; idx += 32) {
    int row = idx >> 3, c = idx & 7;
    ubuf[row * 32 + c] = (_Float16)pa[(r0 + row) * PP + c];
  }

  // ---- per-row scalar work: time encoder + closest-zone argmin -------------
  int cz = 0;
  if (lane < 16) {
    const long row = r0 + lane;
    // time encoder: relu(t*w1+b1) @ w2 + b2 -> abuf cols 128..143 (+ zero 144..159)
    float tv = t_in[row];
    float h[16];
#pragma unroll
    for (int j = 0; j < 16; ++j) h[j] = fmaxf(tv * te_w1[j] + te_b1[j], 0.f);
#pragma unroll
    for (int j = 0; j < 16; ++j) {
      float acc = te_b2[j];
#pragma unroll
      for (int k = 0; k < 16; ++k) acc += h[k] * te_w2[k * 16 + j];
      abuf[lane * 160 + 128 + j] = (_Float16)acc;
    }
#pragma unroll
    for (int j = 0; j < 16; ++j) abuf[lane * 160 + 144 + j] = (_Float16)0.f;

    // argmin_z ( |z|^2 - 2 s.z )  (|s|^2 constant over z)
    float acc[ZZ];
#pragma unroll
    for (int z = 0; z < ZZ; ++z) acc[z] = 0.f;
    const float* sp = state + row * DD;
    for (int d4 = 0; d4 < 16; ++d4) {
      float4 s4 = *(const float4*)(sp + d4 * 4);
#pragma unroll
      for (int z = 0; z < ZZ; ++z) {
        const float* zp = ze + z * DD + d4 * 4;
        acc[z] += s4.x * zp[0] + s4.y * zp[1] + s4.z * zp[2] + s4.w * zp[3];
      }
    }
    float best = 1e30f;
#pragma unroll
    for (int z = 0; z < ZZ; ++z) {
      float d2 = zn2[z] - 2.0f * acc[z];
      if (d2 < best) { best = d2; cz = z; }
    }
  }
  __builtin_amdgcn_wave_barrier();

  // ---- person encoder layer 1: (16x32) @ (32x64) + b, relu -> hbuf ---------
  {
    v16h_t a = lds_load_A(ubuf, 32, 0, lane);
#pragma unroll
    for (int nt = 0; nt < 4; ++nt) {
      v16h_t b = g_load_B(wB, T_PE1 + nt, lane);
      v8f_t c = bias_c(pe_b1, nt * 16, lane);
      c = WMMA(a, b, c);
      store_C_f16(c, hbuf, 64, nt * 16, lane, true);
    }
  }
  __builtin_amdgcn_wave_barrier();

  // ---- person encoder layer 2: (16x64) @ (64x64) + b -> abuf cols 64..127 --
  {
    v16h_t a0 = lds_load_A(hbuf, 64, 0, lane);
    v16h_t a1 = lds_load_A(hbuf, 64, 32, lane);
#pragma unroll
    for (int nt = 0; nt < 4; ++nt) {
      v8f_t c = bias_c(pe_b2, nt * 16, lane);
      c = WMMA(a0, g_load_B(wB, T_PE2 + nt, lane), c);
      c = WMMA(a1, g_load_B(wB, T_PE2 + 4 + nt, lane), c);
      store_C_f16(c, abuf, 160, 64 + nt * 16, lane, false);
    }
  }
  __builtin_amdgcn_wave_barrier();

  // ---- flow net layer 1: (16x160) @ (160x64) + b, relu -> hbuf -------------
  {
    v16h_t a[5];
#pragma unroll
    for (int kc = 0; kc < 5; ++kc) a[kc] = lds_load_A(abuf, 160, kc * 32, lane);
#pragma unroll
    for (int nt = 0; nt < 4; ++nt) {
      v8f_t c = bias_c(fn_b1, nt * 16, lane);
#pragma unroll
      for (int kc = 0; kc < 5; ++kc)
        c = WMMA(a[kc], g_load_B(wB, T_FN1 + kc * 4 + nt, lane), c);
      store_C_f16(c, hbuf, 64, nt * 16, lane, true);
    }
  }
  __builtin_amdgcn_wave_barrier();

  // ---- flow net layer 2: (16x64) @ (64x32) + b, relu -> ubuf (16x32) -------
  {
    v16h_t a0 = lds_load_A(hbuf, 64, 0, lane);
    v16h_t a1 = lds_load_A(hbuf, 64, 32, lane);
#pragma unroll
    for (int nt = 0; nt < 2; ++nt) {
      v8f_t c = bias_c(fn_b2, nt * 16, lane);
      c = WMMA(a0, g_load_B(wB, T_FN2 + nt, lane), c);
      c = WMMA(a1, g_load_B(wB, T_FN2 + 2 + nt, lane), c);
      store_C_f16(c, ubuf, 32, nt * 16, lane, true);
    }
  }
  __builtin_amdgcn_wave_barrier();

  // ---- flow net layer 3: (16x32) @ (32x64) + b -> v in ubuf (16x64) --------
  {
    v16h_t a = lds_load_A(ubuf, 32, 0, lane);
    v8f_t cs[4];
#pragma unroll
    for (int nt = 0; nt < 4; ++nt) {
      cs[nt] = bias_c(fn_b3, nt * 16, lane);
      cs[nt] = WMMA(a, g_load_B(wB, T_FN3 + nt, lane), cs[nt]);
    }
#pragma unroll
    for (int nt = 0; nt < 4; ++nt)
      store_C_f16(cs[nt], ubuf, 64, nt * 16, lane, false);
  }
  __builtin_amdgcn_wave_barrier();

  // ---- masked softmax over adjacent directions, blend, store ---------------
  if (lane < 16) {
    const long row = r0 + lane;
    const float* dz = dir + cz * ZZ * DD;
    const _Float16* vrow = ubuf + lane * 64;

    float proj[ZZ];
#pragma unroll
    for (int z = 0; z < ZZ; ++z) proj[z] = 0.f;
    for (int d = 0; d < DD; ++d) {
      float vd = (float)vrow[d];
#pragma unroll
      for (int z = 0; z < ZZ; ++z) proj[z] += dz[z * DD + d] * vd;
    }

    unsigned mb = adjbits[cz];
    float mx = -3.0e38f;
#pragma unroll
    for (int z = 0; z < ZZ; ++z)
      if ((mb >> z) & 1u) mx = fmaxf(mx, proj[z]);
    float w[ZZ], s = 0.f;
#pragma unroll
    for (int z = 0; z < ZZ; ++z) {
      float e = ((mb >> z) & 1u) ? __expf(proj[z] - mx) : 0.f;
      w[z] = e; s += e;
    }
    float inv = 0.5f / s;

    for (int d4 = 0; d4 < 16; ++d4) {
      float4 o;
      float a0 = 0.f, a1 = 0.f, a2 = 0.f, a3 = 0.f;
#pragma unroll
      for (int z = 0; z < ZZ; ++z) {
        const float* dp = dz + z * DD + d4 * 4;
        a0 += w[z] * dp[0]; a1 += w[z] * dp[1];
        a2 += w[z] * dp[2]; a3 += w[z] * dp[3];
      }
      o.x = a0 * inv; o.y = a1 * inv; o.z = a2 * inv; o.w = a3 * inv;
      *(float4*)(out + row * DD + d4 * 4) = o;
    }
  }
}

// ---------------------------------------------------------------------------
// Launcher
// ---------------------------------------------------------------------------
extern "C" void kernel_launch(void* const* d_in, const int* in_sizes, int n_in,
                              void* d_out, int out_size, void* d_ws, size_t ws_size,
                              hipStream_t stream) {
  const float* t_in   = (const float*)d_in[0];
  const float* state  = (const float*)d_in[1];
  const float* pa     = (const float*)d_in[2];
  const int*   edge   = (const int*)d_in[3];
  const float* ze     = (const float*)d_in[4];
  const float* te_w1  = (const float*)d_in[5];
  const float* te_b1  = (const float*)d_in[6];
  const float* te_w2  = (const float*)d_in[7];
  const float* te_b2  = (const float*)d_in[8];
  const float* pe_w1  = (const float*)d_in[9];
  const float* pe_b1  = (const float*)d_in[10];
  const float* pe_w2  = (const float*)d_in[11];
  const float* pe_b2  = (const float*)d_in[12];
  const float* fn_w1  = (const float*)d_in[13];
  const float* fn_b1  = (const float*)d_in[14];
  const float* fn_w2  = (const float*)d_in[15];
  const float* fn_b2  = (const float*)d_in[16];
  const float* fn_w3  = (const float*)d_in[17];
  const float* fn_b3  = (const float*)d_in[18];
  float* out = (float*)d_out;

  char* ws = (char*)d_ws;
  float*     dir     = (float*)(ws);                 // 8*8*64 f32 = 16384 B
  unsigned*  adjbits = (unsigned*)(ws + 16384);      // 32 B
  float*     zn2     = (float*)(ws + 16416);         // 32 B
  _Float16*  wB      = (_Float16*)(ws + 16448);      // 40*512*2 = 40960 B

  ode_prep<<<1, 256, 0, stream>>>(edge, ze, pe_w1, pe_w2, fn_w1, fn_w2, fn_w3,
                                  dir, adjbits, zn2, wB);

  const int B = in_sizes[0];                         // 262144
  dim3 grid(B / (WAVES * 16));
  ode_main<<<grid, WAVES * 32, 0, stream>>>(
      t_in, state, pa, ze, te_w1, te_b1, te_w2, te_b2,
      pe_b1, pe_b2, fn_b1, fn_b2, fn_b3,
      dir, adjbits, zn2, wB, out);
}
